// DualLaplacianBlock_32590211842338
// MI455X (gfx1250) — matrix operand order repaired and checked by reading, not compiled, not run on hardware
//
#include <hip/hip_runtime.h>
#include <hip/hip_bf16.h>
#include <math.h>

// ---------- types ----------
typedef __attribute__((ext_vector_type(8)))  unsigned short v8u16;
typedef __attribute__((ext_vector_type(16))) unsigned short v16u16;
typedef __attribute__((ext_vector_type(16))) __bf16         v16bf;
typedef __attribute__((ext_vector_type(8)))  float          v8f;

#define EPSF 1e-8f

__device__ __forceinline__ unsigned short f2bf(float f) {
  union { float f; unsigned u; } v; v.f = f;
  unsigned r = v.u + 0x7fffu + ((v.u >> 16) & 1u);   // RNE
  return (unsigned short)(r >> 16);
}
__device__ __forceinline__ float bf2f(unsigned short u) {
  union { unsigned u; float f; } v; v.u = ((unsigned)u) << 16;
  return v.f;
}

// A-matrix 16x32 bf16 fragment (lane L holds row M=L%16;
// lanes<16 hold K=k0..k0+15 contiguous, lanes>=16 hold K=k0+16..k0+31).
__device__ __forceinline__ v16bf fragA(const unsigned short* __restrict__ row, int k0, int hi) {
  v16u16 t = *(const v16u16*)(row + k0 + hi * 16);
  return __builtin_bit_cast(v16bf, t);
}
// B-matrix 32x16 bf16 fragment (lane L holds column N=L%16; lanes<16 hold
// K=k0..k0+15 contiguous, lanes>=16 hold K=k0+16..k0+31).
__device__ __forceinline__ v16bf fragB(const unsigned short* __restrict__ row, int k0, int hi) {
  v16u16 t = *(const v16u16*)(row + k0 + hi * 16);
  return __builtin_bit_cast(v16bf, t);
}

#define WMMA_BF16(a, b, c) \
  __builtin_amdgcn_wmma_f32_16x16x32_bf16(false, (a), false, (b), (short)0, (c), false, false)

// ---------- fp32 -> bf16 convert ----------
__global__ void k_f32_to_bf16(const float* __restrict__ src, unsigned short* __restrict__ dst, long n) {
  long i = (long)blockIdx.x * blockDim.x + threadIdx.x;
  if (i < n) dst[i] = f2bf(src[i]);
}

// ---------- generic WMMA GEMM: C[b] = A[b] (MxK, row-major) * B[b]^T (NcxK, row-major) ----------
// One wave computes a 32x32 output tile: 2 A-frags x 2 B-frags -> 4 WMMAs per K-step,
// halving bytes/WMMA vs 16x16 and giving 4 independent accumulators to hide WMMA latency.
template <bool OUT_BF16>
__global__ void k_gemm_tn(const unsigned short* __restrict__ A,
                          const unsigned short* __restrict__ Bm,
                          void* __restrict__ Cv,
                          int Nc, int K,
                          long sA, long sB, long sC,
                          int tilesM, int tilesN, int nBatch) {   // tiles of 32x32
  int wave = (int)((blockIdx.x * blockDim.x + threadIdx.x) >> 5);
  int lane = threadIdx.x & 31;
  int tilesPerBatch = tilesM * tilesN;
  if (wave >= tilesPerBatch * nBatch) return;          // wave-uniform exit
  int b  = wave / tilesPerBatch;
  int t  = wave - b * tilesPerBatch;
  int tm = t / tilesN;
  int tn = t - tm * tilesN;

  int hi = lane >> 4, lo = lane & 15;
  const unsigned short* a0p = A + (long)b * sA + (long)(tm * 32 + lo) * K;
  const unsigned short* a1p = a0p + (long)16 * K;
  const unsigned short* b0p = Bm + (long)b * sB + (long)(tn * 32 + lo) * K;
  const unsigned short* b1p = b0p + (long)16 * K;

  v8f c00 = {}, c01 = {}, c10 = {}, c11 = {};
  for (int k0 = 0; k0 < K; k0 += 32) {
    __builtin_prefetch(a0p + k0 + 512, 0, 1);          // global_prefetch_b8
    __builtin_prefetch(b0p + k0 + 512, 0, 1);
    v16bf a0 = fragA(a0p, k0, hi);
    v16bf a1 = fragA(a1p, k0, hi);
    v16bf w0 = fragB(b0p, k0, hi);
    v16bf w1 = fragB(b1p, k0, hi);
    c00 = WMMA_BF16(a0, w0, c00);
    c01 = WMMA_BF16(a0, w1, c01);
    c10 = WMMA_BF16(a1, w0, c10);
    c11 = WMMA_BF16(a1, w1, c11);
  }

  int row0 = tm * 32 + 8 * hi;
  int col0 = tn * 32 + lo;
  if (OUT_BF16) {
    unsigned short* C = (unsigned short*)Cv + (long)b * sC;
#pragma unroll
    for (int r = 0; r < 8; ++r) {
      C[(long)(row0 + r) * Nc + col0]           = f2bf(c00[r]);
      C[(long)(row0 + r) * Nc + col0 + 16]      = f2bf(c01[r]);
      C[(long)(row0 + r + 16) * Nc + col0]      = f2bf(c10[r]);
      C[(long)(row0 + r + 16) * Nc + col0 + 16] = f2bf(c11[r]);
    }
  } else {
    float* C = (float*)Cv + (long)b * sC;
#pragma unroll
    for (int r = 0; r < 8; ++r) {
      C[(long)(row0 + r) * Nc + col0]           = c00[r];
      C[(long)(row0 + r) * Nc + col0 + 16]      = c01[r];
      C[(long)(row0 + r + 16) * Nc + col0]      = c10[r];
      C[(long)(row0 + r + 16) * Nc + col0 + 16] = c11[r];
    }
  }
}

// ---------- per-row stats: invnorm(z_l), sqnorm(z_g) ----------
__global__ void k_rowstats(const unsigned short* __restrict__ zl,
                           const unsigned short* __restrict__ zg,
                           float* __restrict__ invn, float* __restrict__ sqg, int D) {
  int row = blockIdx.x;
  const unsigned short* pl = zl + (long)row * D;
  const unsigned short* pg = zg + (long)row * D;
  float sl = 0.f, sg = 0.f;
  for (int i = threadIdx.x; i < D; i += blockDim.x) {
    float a = bf2f(pl[i]); float g = bf2f(pg[i]);
    sl += a * a; sg += g * g;
  }
  __shared__ float s1[256], s2[256];
  s1[threadIdx.x] = sl; s2[threadIdx.x] = sg;
  __syncthreads();
  for (int s = blockDim.x >> 1; s > 0; s >>= 1) {
    if (threadIdx.x < (unsigned)s) { s1[threadIdx.x] += s1[threadIdx.x + s]; s2[threadIdx.x] += s2[threadIdx.x + s]; }
    __syncthreads();
  }
  if (threadIdx.x == 0) {
    invn[row] = 1.0f / fmaxf(sqrtf(s1[0]), EPSF);
    sqg[row]  = s2[0];
  }
}

// ---------- pass 1: causal row sums of A_l and A_g (one wave per query tile) ----------
__global__ void k_attn_rowsum(const unsigned short* __restrict__ zl,
                              const unsigned short* __restrict__ zg,
                              const float* __restrict__ invn, const float* __restrict__ sqg,
                              float* __restrict__ rslO, float* __restrict__ rsgO,
                              const float* __restrict__ lsig,
                              int N, int D, int nBatch) {
  int wave = (int)((blockIdx.x * blockDim.x + threadIdx.x) >> 5);
  int lane = threadIdx.x & 31;
  int QT = N >> 4;
  int qt = wave % QT;
  int b  = wave / QT;
  if (b >= nBatch) return;

  float sigma = __expf(lsig[0]);
  float inv2s2 = 1.0f / (2.0f * sigma * sigma);

  int hi = lane >> 4, lo = lane & 15;
  long base = (long)b * N;
  const unsigned short* zlb = zl + base * D;
  const unsigned short* zgb = zg + base * D;
  const unsigned short* qla = zlb + (long)(qt * 16 + lo) * D;
  const unsigned short* qga = zgb + (long)(qt * 16 + lo) * D;

  float invq[8], sqq[8];
#pragma unroll
  for (int r = 0; r < 8; ++r) {
    int m = qt * 16 + r + 8 * hi;
    invq[r] = invn[base + m]; sqq[r] = sqg[base + m];
  }
  float rsl[8] = {}, rsg[8] = {};

  for (int kt = 0; kt <= qt; ++kt) {
    const unsigned short* kla = zlb + (long)(kt * 16 + lo) * D;
    const unsigned short* kga = zgb + (long)(kt * 16 + lo) * D;
    v8f Sl = {}, Sg = {};
    for (int k0 = 0; k0 < D; k0 += 32) {
      Sl = WMMA_BF16(fragA(qla, k0, hi), fragB(kla, k0, hi), Sl);
      Sg = WMMA_BF16(fragA(qga, k0, hi), fragB(kga, k0, hi), Sg);
    }
    float invk = invn[base + kt * 16 + lo];
    float sqk  = sqg[base + kt * 16 + lo];
#pragma unroll
    for (int r = 0; r < 8; ++r) {
      int M = qt * 16 + r + 8 * hi;
      int C = kt * 16 + lo;
      bool keep = (C < M);                              // tril minus diagonal
      float al = keep ? fmaxf(Sl[r], 0.f) * invq[r] * invk : 0.f;
      float d2 = fmaxf(sqq[r] + sqk - 2.f * Sg[r], 0.f);
      float ag = keep ? __expf(-d2 * inv2s2) : 0.f;
      rsl[r] += al; rsg[r] += ag;
    }
  }
#pragma unroll
  for (int r = 0; r < 8; ++r)
    for (int m = 1; m < 16; m <<= 1) {
      rsl[r] += __shfl_xor(rsl[r], m);
      rsg[r] += __shfl_xor(rsg[r], m);
    }
  if (lo == 0) {
#pragma unroll
    for (int r = 0; r < 8; ++r) {
      int M = qt * 16 + r + 8 * hi;
      rslO[base + M] = rsl[r];
      rsgO[base + M] = rsg[r];
    }
  }
}

// ---------- pass 2: materialize gated, row-normalized K (bf16), zero upper tiles ----------
__global__ void k_attn_kmat(const unsigned short* __restrict__ zl,
                            const unsigned short* __restrict__ zg,
                            const float* __restrict__ invn, const float* __restrict__ sqg,
                            const float* __restrict__ rsl, const float* __restrict__ rsg,
                            unsigned short* __restrict__ Kout,
                            const float* __restrict__ gate, const float* __restrict__ lsig,
                            int N, int D, int nBatch) {
  int wave = (int)((blockIdx.x * blockDim.x + threadIdx.x) >> 5);
  int lane = threadIdx.x & 31;
  int QT = N >> 4;
  int kt = wave % QT;
  int tmp = wave / QT;
  int qt = tmp % QT;
  int b  = tmp / QT;
  if (b >= nBatch) return;

  int hi = lane >> 4, lo = lane & 15;
  unsigned short* Kb = Kout + (long)b * N * N;
  int col = kt * 16 + lo;

  if (kt > qt) {                                        // strictly upper tile: zero
#pragma unroll
    for (int r = 0; r < 8; ++r)
      Kb[(long)(qt * 16 + r + 8 * hi) * N + col] = 0;
    return;
  }

  float sigma = __expf(lsig[0]);
  float inv2s2 = 1.0f / (2.0f * sigma * sigma);
  float wl = 1.0f / (1.0f + __expf(-gate[0]));
  float wg = 1.0f - wl;

  long base = (long)b * N;
  const unsigned short* zlb = zl + base * D;
  const unsigned short* zgb = zg + base * D;
  const unsigned short* qla = zlb + (long)(qt * 16 + lo) * D;
  const unsigned short* qga = zgb + (long)(qt * 16 + lo) * D;
  const unsigned short* kla = zlb + (long)(kt * 16 + lo) * D;
  const unsigned short* kga = zgb + (long)(kt * 16 + lo) * D;

  v8f Sl = {}, Sg = {};
  for (int k0 = 0; k0 < D; k0 += 32) {
    Sl = WMMA_BF16(fragA(qla, k0, hi), fragB(kla, k0, hi), Sl);
    Sg = WMMA_BF16(fragA(qga, k0, hi), fragB(kga, k0, hi), Sg);
  }
  float invk = invn[base + kt * 16 + lo];
  float sqk  = sqg[base + kt * 16 + lo];
#pragma unroll
  for (int r = 0; r < 8; ++r) {
    int M = qt * 16 + r + 8 * hi;
    bool keep = (col < M);
    float al = keep ? fmaxf(Sl[r], 0.f) * invn[base + M] * invk : 0.f;
    float d2 = fmaxf(sqg[base + M] + sqk - 2.f * Sg[r], 0.f);
    float ag = keep ? __expf(-d2 * inv2s2) : 0.f;
    float kv = wl * al / fmaxf(rsl[base + M], EPSF) + wg * ag / fmaxf(rsg[base + M], EPSF);
    Kb[(long)M * N + col] = f2bf(kv);
  }
}

// ---------- transpose via async global->LDS copies (ASYNCcnt path) ----------
// Per CDNA5 ISA: GLOBAL_LOAD_ASYNC_TO_LDS_B32 takes a per-lane LDS byte
// address in VDST and a per-lane 64-bit global address; tracked by ASYNCcnt.
// Per §10.2, the low 32 bits of a generic LDS pointer are the LDS byte offset.
__device__ __forceinline__ void async_g2lds_b32(unsigned ldsAddr, const void* gaddr) {
  asm volatile("global_load_async_to_lds_b32 %0, %1, off"
               :: "v"(ldsAddr), "v"(gaddr)
               : "memory");
}
__device__ __forceinline__ void wait_asynccnt0() {
  asm volatile("s_wait_asynccnt 0x0" ::: "memory");
}

// src [b][rows][cols] -> dst [b][cols][rows], bf16, 32x32 tiles, block (16,16)
__global__ void k_transpose_bf16(const unsigned short* __restrict__ src,
                                 unsigned short* __restrict__ dst, int rows, int cols) {
  __shared__ unsigned short tile[32][34];               // +2 shorts pad, keeps 4B align
  int b  = blockIdx.z;
  int c0 = blockIdx.x * 32, r0 = blockIdx.y * 32;
  const unsigned short* s = src + (long)b * rows * cols;
  unsigned short* d = dst + (long)b * rows * cols;
  int tx = threadIdx.x, ty = threadIdx.y;               // 16 x 16

  const unsigned short* sp0 = s + (long)(r0 + ty) * cols + c0 + tx * 2;
  const unsigned short* sp1 = s + (long)(r0 + ty + 16) * cols + c0 + tx * 2;
  unsigned l0 = (unsigned)(unsigned long long)&tile[ty][tx * 2];
  unsigned l1 = (unsigned)(unsigned long long)&tile[ty + 16][tx * 2];
  async_g2lds_b32(l0, sp0);                             // global_load_async_to_lds_b32
  async_g2lds_b32(l1, sp1);
  wait_asynccnt0();                                     // s_wait_asynccnt 0
  __syncthreads();

  unsigned p0 = (unsigned)tile[tx * 2][ty]      | ((unsigned)tile[tx * 2 + 1][ty] << 16);
  unsigned p1 = (unsigned)tile[tx * 2][ty + 16] | ((unsigned)tile[tx * 2 + 1][ty + 16] << 16);
  *(unsigned*)(d + (long)(c0 + ty) * rows + r0 + tx * 2)      = p0;
  *(unsigned*)(d + (long)(c0 + ty + 16) * rows + r0 + tx * 2) = p1;
}

// ---------- host launcher ----------
extern "C" void kernel_launch(void* const* d_in, const int* in_sizes, int n_in,
                              void* d_out, int out_size, void* d_ws, size_t ws_size,
                              hipStream_t stream) {
  (void)in_sizes; (void)n_in; (void)out_size; (void)ws_size;
  const int B = 4, N = 2048, D = 1024;
  const long BN = (long)B * N;

  const float* h    = (const float*)d_in[0];
  // d_in[1] = causal_mask (implicit in kernels)
  const float* Wl   = (const float*)d_in[2];
  const float* Wg   = (const float*)d_in[3];
  const float* Wv   = (const float*)d_in[4];
  const float* Wo   = (const float*)d_in[5];
  const float* gate = (const float*)d_in[6];
  const float* lsig = (const float*)d_in[7];

  char* ws = (char*)d_ws;
  size_t off = 0;
  auto alloc = [&](size_t bytes) -> void* {
    void* p = ws + off;
    off = (off + bytes + 255) & ~(size_t)255;
    return p;
  };
  unsigned short* h_bf   = (unsigned short*)alloc((size_t)BN * D * 2);
  unsigned short* Wl_bf  = (unsigned short*)alloc((size_t)D * D * 2);
  unsigned short* Wg_bf  = (unsigned short*)alloc((size_t)D * D * 2);
  unsigned short* Wv_bf  = (unsigned short*)alloc((size_t)D * D * 2);
  unsigned short* Wo_bf  = (unsigned short*)alloc((size_t)D * D * 2);
  unsigned short* zl_bf  = (unsigned short*)alloc((size_t)BN * D * 2);
  unsigned short* zg_bf  = (unsigned short*)alloc((size_t)BN * D * 2);
  unsigned short* v_bf   = (unsigned short*)alloc((size_t)BN * D * 2);
  unsigned short* vt_bf  = (unsigned short*)alloc((size_t)BN * D * 2);
  unsigned short* at_bf  = (unsigned short*)alloc((size_t)BN * D * 2);
  unsigned short* Kmat   = (unsigned short*)alloc((size_t)B * N * N * 2);
  float* invn = (float*)alloc((size_t)BN * 4);
  float* sqg  = (float*)alloc((size_t)BN * 4);
  float* rsl  = (float*)alloc((size_t)BN * 4);
  float* rsg  = (float*)alloc((size_t)BN * 4);

  auto cdiv = [](long a, long b) { return (int)((a + b - 1) / b); };

  // 1) fp32 -> bf16
  k_f32_to_bf16<<<cdiv(BN * D, 256), 256, 0, stream>>>(h, h_bf, BN * D);
  k_f32_to_bf16<<<cdiv((long)D * D, 256), 256, 0, stream>>>(Wl, Wl_bf, (long)D * D);
  k_f32_to_bf16<<<cdiv((long)D * D, 256), 256, 0, stream>>>(Wg, Wg_bf, (long)D * D);
  k_f32_to_bf16<<<cdiv((long)D * D, 256), 256, 0, stream>>>(Wv, Wv_bf, (long)D * D);
  k_f32_to_bf16<<<cdiv((long)D * D, 256), 256, 0, stream>>>(Wo, Wo_bf, (long)D * D);

  // 2) projections: z = h @ W^T  (M=BN, Nc=D, K=D), 32x32 tiles
  int tM = (int)(BN / 32), tN = D / 32;
  int projBlocks = cdiv((long)tM * tN * 32, 256);
  k_gemm_tn<true><<<projBlocks, 256, 0, stream>>>(h_bf, Wl_bf, zl_bf, D, D, 0, 0, 0, tM, tN, 1);
  k_gemm_tn<true><<<projBlocks, 256, 0, stream>>>(h_bf, Wg_bf, zg_bf, D, D, 0, 0, 0, tM, tN, 1);
  k_gemm_tn<true><<<projBlocks, 256, 0, stream>>>(h_bf, Wv_bf, v_bf, D, D, 0, 0, 0, tM, tN, 1);

  // 3) row stats
  k_rowstats<<<(int)BN, 256, 0, stream>>>(zl_bf, zg_bf, invn, sqg, D);

  // 4) causal row sums
  int QT = N / 16;
  k_attn_rowsum<<<cdiv((long)B * QT * 32, 256), 256, 0, stream>>>(
      zl_bf, zg_bf, invn, sqg, rsl, rsg, lsig, N, D, B);

  // 5) materialize K (bf16, row-normalized, gated, zero above diagonal)
  k_attn_kmat<<<cdiv((long)B * QT * QT * 32, 256), 256, 0, stream>>>(
      zl_bf, zg_bf, invn, sqg, rsl, rsg, Kmat, gate, lsig, N, D, B);

  // 6) transpose v -> vt (async global->LDS path)
  k_transpose_bf16<<<dim3(D / 32, N / 32, B), dim3(16, 16), 0, stream>>>(v_bf, vt_bf, N, D);

  // 7) attn_out[b] = K[b] @ v[b]   (M=N, Nc=D, K=N), batched, 32x32 tiles
  int avBlocks = cdiv((long)B * (N / 32) * (D / 32) * 32, 256);
  k_gemm_tn<true><<<avBlocks, 256, 0, stream>>>(Kmat, vt_bf, at_bf, D, N,
                                                (long)N * N, (long)D * N, (long)N * D,
                                                N / 32, D / 32, B);

  // 8) out = attn_out @ W_O^T -> fp32 d_out
  k_gemm_tn<false><<<projBlocks, 256, 0, stream>>>(at_bf, Wo_bf, d_out, D, D, 0, 0, 0, tM, tN, 1);
}
